// AttEGNNLayer_84138409328670
// MI455X (gfx1250) — matrix operand order repaired and checked by reading, not compiled, not run on hardware
//
#include <hip/hip_runtime.h>
#include <hip/hip_bf16.h>
#include <math.h>

// ---------------- problem constants (match setup_inputs) ----------------
constexpr int BB = 2;       // batch
constexpr int NN = 2048;    // nodes
constexpr int NE = 2048;    // edges
constexpr int DD = 128;     // node dim
constexpr int ED = 64;      // edge dim
constexpr int HD = 64;      // hidden / head dim
constexpr int NH = 4;       // heads
constexpr float LN_EPS = 1e-3f;

typedef __attribute__((ext_vector_type(16))) __bf16 bf16x16;
typedef __attribute__((ext_vector_type(8)))  float  f32x8;

// ---------------- WMMA helpers ----------------
__device__ __forceinline__ f32x8 wmma_bf16(bf16x16 a, bf16x16 b, f32x8 c) {
  return __builtin_amdgcn_wmma_f32_16x16x32_bf16(false, a, false, b, (short)0, c, false, false);
}

__device__ __forceinline__ int frag_klo(int lane) { return (lane & 16) ? 8 : 0; }

// A-fragment (16x32, row-major source, lane's row pointer given).
// lane half 0: K = {klo..klo+7, klo+16..klo+23} with klo=0 ; half 1: klo=8.
__device__ __forceinline__ bf16x16 ldfrag_bf16(const __bf16* p_row, int k0, int lane) {
  const __bf16* p = p_row + k0 + frag_klo(lane);
  bf16x16 a;
#pragma unroll
  for (int e = 0; e < 8; ++e) a[e] = p[e];
#pragma unroll
  for (int e = 0; e < 8; ++e) a[8 + e] = p[16 + e];
  return a;
}

__device__ __forceinline__ bf16x16 ldfrag_f32(const float* p_row, int k0, int lane) {
  const float* p = p_row + k0 + frag_klo(lane);
  bf16x16 a;
#pragma unroll
  for (int e = 0; e < 8; ++e) a[e] = (__bf16)p[e];
#pragma unroll
  for (int e = 0; e < 8; ++e) a[8 + e] = (__bf16)p[16 + e];
  return a;
}

// A-fragment from strided f32 (e.g. transposed access of incidence): element K=k at base[k*stride]
__device__ __forceinline__ bf16x16 ldfrag_f32_strided(const float* base, long stride, int k0, int lane) {
  int klo = frag_klo(lane);
  bf16x16 a;
#pragma unroll
  for (int e = 0; e < 8; ++e) a[e] = (__bf16)base[(long)(k0 + klo + e) * stride];
#pragma unroll
  for (int e = 0; e < 8; ++e) a[8 + e] = (__bf16)base[(long)(k0 + 16 + klo + e) * stride];
  return a;
}

// B-fragment (32x16): source stored row-major [Ncol][K]; lane holds col = lane&15,
// half 0: K = {k0..k0+7, k0+16..k0+23}; half 1: K = {k0+8..k0+15, k0+24..k0+31}.
__device__ __forceinline__ bf16x16 ldfragB_bf16(const __bf16* p_col, int k0, int lane) {
  const __bf16* p = p_col + k0 + frag_klo(lane);
  bf16x16 a;
#pragma unroll
  for (int e = 0; e < 8; ++e) a[e] = p[e];
#pragma unroll
  for (int e = 0; e < 8; ++e) a[8 + e] = p[16 + e];
  return a;
}

// ---------------- small utility kernels ----------------
__global__ void k_cvt(__bf16* dst, const float* src, int n) {
  int i = blockIdx.x * 256 + threadIdx.x;
  if (i < n) dst[i] = (__bf16)src[i];
}

// dst[c*Rpad + r] = (r < R) ? src[r*C + c] : 0
__global__ void k_transpose(__bf16* dst, const float* src, int R, int C, int Rpad) {
  int i = blockIdx.x * 256 + threadIdx.x;
  if (i >= C * Rpad) return;
  int c = i / Rpad, r = i % Rpad;
  dst[i] = (r < R) ? (__bf16)src[r * C + c] : (__bf16)0.0f;
}

__global__ void k_bias(float* bb, const float* x, const float* node_bias) {
  int i = blockIdx.x * 256 + threadIdx.x;
  if (i >= NH * BB * NN) return;
  int h = i / (BB * NN);
  long bn = i % (BB * NN);
  const float* xr = x + bn * DD;
  const float* w = node_bias + h * DD;
  float s = 0.f;
  for (int d = 0; d < DD; ++d) s += xr[d] * w[d];
  bb[i] = s;
}

// ---------------- projections: Q,K,V,G ----------------
__global__ __launch_bounds__(128) void k_proj(const __bf16* xb, const __bf16* WqT, const __bf16* WkT,
                       const __bf16* WvT, const __bf16* WgT,
                       __bf16* Qb, __bf16* Kb, __bf16* Vt, __bf16* Gb) {
  int n0 = blockIdx.x * 16, b = blockIdx.y, h = blockIdx.z;
  int lane = threadIdx.x & 31, wv = threadIdx.x >> 5;
  int c = lane & 15, mb = 8 * (lane >> 4);
  const __bf16* A = xb + ((long)b * NN + n0 + c) * DD;
  const __bf16* wq = WqT + (long)h * HD * DD;
  const __bf16* wk = WkT + (long)h * HD * DD;
  const __bf16* wvv = WvT + (long)h * DD * DD;
  const __bf16* wg = WgT + (long)h * DD * DD;
  int jq = wv * 16, jv = wv * 32;
  f32x8 aq{}, ak{}, av0{}, av1{}, ag0{}, ag1{};
  for (int kt = 0; kt < DD / 32; ++kt) {
    int k0 = kt * 32;
    bf16x16 a = ldfrag_bf16(A, k0, lane);
    aq  = wmma_bf16(a, ldfragB_bf16(wq  + (long)(jq + c) * DD, k0, lane), aq);
    ak  = wmma_bf16(a, ldfragB_bf16(wk  + (long)(jq + c) * DD, k0, lane), ak);
    av0 = wmma_bf16(a, ldfragB_bf16(wvv + (long)(jv + c) * DD, k0, lane), av0);
    av1 = wmma_bf16(a, ldfragB_bf16(wvv + (long)(jv + 16 + c) * DD, k0, lane), av1);
    ag0 = wmma_bf16(a, ldfragB_bf16(wg  + (long)(jv + c) * DD, k0, lane), ag0);
    ag1 = wmma_bf16(a, ldfragB_bf16(wg  + (long)(jv + 16 + c) * DD, k0, lane), ag1);
  }
  long hb = (long)h * BB + b;
#pragma unroll
  for (int r = 0; r < 8; ++r) {
    int m = n0 + mb + r;
    Qb[(hb * NN + m) * HD + jq + c] = (__bf16)aq[r];
    Kb[(hb * NN + m) * HD + jq + c] = (__bf16)ak[r];
    Vt[(hb * DD + jv + c) * NN + m] = (__bf16)av0[r];
    Vt[(hb * DD + jv + 16 + c) * NN + m] = (__bf16)av1[r];
    Gb[(hb * NN + m) * DD + jv + c] = (__bf16)(1.f / (1.f + __expf(-ag0[r])));
    Gb[(hb * NN + m) * DD + jv + 16 + c] = (__bf16)(1.f / (1.f + __expf(-ag1[r])));
  }
}

// ---------------- fused attention: logits(LDS) -> softmax -> P@V -> +x -> LN1 -> *G ----------------
__global__ __launch_bounds__(128) void k_attn(const __bf16* Qb, const __bf16* Kb, const __bf16* Vt,
                       const __bf16* Gb, const float* Bbuf, const float* node_adj,
                       const float* x, const float* ln1g, const float* ln1b,
                       __bf16* attn_out) {
  int n0 = blockIdx.x * 16, b = blockIdx.y, h = blockIdx.z;
  int tid = threadIdx.x, lane = tid & 31, wv = tid >> 5;
  int c = lane & 15, mb = 8 * (lane >> 4);
  long hb = (long)h * BB + b;
  __shared__ __bf16 slog[16][NN];        // 64 KB: logits then masked probs
  __shared__ float spart[4][16];
  __shared__ float ssump[16][8];
  __shared__ float sstat[16];
  __shared__ float ssum[16];
  __shared__ float otile[16][DD];
  const float scale = 0.125f;  // 1/sqrt(64)
  const __bf16* qrow = Qb + (hb * NN + n0 + c) * HD;
  bf16x16 qa0 = ldfrag_bf16(qrow, 0, lane);
  bf16x16 qa1 = ldfrag_bf16(qrow, 32, lane);
  float bbv[8]; const float* adjrow[8]; float rmax[8];
#pragma unroll
  for (int r = 0; r < 8; ++r) {
    bbv[r] = Bbuf[hb * NN + n0 + mb + r];
    adjrow[r] = node_adj + ((long)b * NN + n0 + mb + r) * NN;
    rmax[r] = -3.0e38f;
  }
  // pass 1: masked+biased logits -> LDS, track per-row max
  for (int mt = wv; mt < NN / 16; mt += 4) {
    int m0 = mt * 16;
    const __bf16* krow = Kb + (hb * NN + m0 + c) * HD;
    f32x8 acc{};
    acc = wmma_bf16(qa0, ldfragB_bf16(krow, 0, lane), acc);
    acc = wmma_bf16(qa1, ldfragB_bf16(krow, 32, lane), acc);
#pragma unroll
    for (int r = 0; r < 8; ++r) {
      float adj = adjrow[r][m0 + c];
      float l = acc[r] * scale + bbv[r] - 10000.0f * (1.0f - adj);
      slog[mb + r][m0 + c] = (__bf16)l;
      rmax[r] = fmaxf(rmax[r], l);
    }
  }
#pragma unroll
  for (int r = 0; r < 8; ++r) {
    float m = rmax[r];
    m = fmaxf(m, __shfl_xor(m, 1));
    m = fmaxf(m, __shfl_xor(m, 2));
    m = fmaxf(m, __shfl_xor(m, 4));
    m = fmaxf(m, __shfl_xor(m, 8));
    rmax[r] = m;
  }
  if (c == 0) {
#pragma unroll
    for (int r = 0; r < 8; ++r) spart[wv][mb + r] = rmax[r];
  }
  __syncthreads();
  if (tid < 16)
    sstat[tid] = fmaxf(fmaxf(spart[0][tid], spart[1][tid]), fmaxf(spart[2][tid], spart[3][tid]));
  __syncthreads();
  // pass 2: exp, global sum (pre-mask), store masked probs in place
  {
    int row = tid >> 3, c0 = tid & 7;
    float mrow = sstat[row];
    const float* arow = node_adj + ((long)b * NN + n0 + row) * NN;
    float part = 0.f;
    for (int m = c0; m < NN; m += 8) {
      float p = __expf((float)slog[row][m] - mrow);
      part += p;
      slog[row][m] = (__bf16)(p * arow[m]);
    }
    ssump[row][c0] = part;
  }
  __syncthreads();
  if (tid < 16) {
    float s = 0.f;
#pragma unroll
    for (int j = 0; j < 8; ++j) s += ssump[tid][j];
    ssum[tid] = s;
  }
  __syncthreads();
  // pass 3: O = P@V (rescale by 1/sum at epilogue), residual +x
  {
    int d0 = wv * 32;
    const __bf16* arow = &slog[c][0];
    const __bf16* v0 = Vt + (hb * DD + d0 + c) * NN;
    const __bf16* v1 = Vt + (hb * DD + d0 + 16 + c) * NN;
    f32x8 acc0{}, acc1{};
    for (int kt = 0; kt < NN / 32; ++kt) {
      int k0 = kt * 32;
      bf16x16 a = ldfrag_bf16(arow, k0, lane);
      acc0 = wmma_bf16(a, ldfragB_bf16(v0, k0, lane), acc0);
      acc1 = wmma_bf16(a, ldfragB_bf16(v1, k0, lane), acc1);
    }
#pragma unroll
    for (int r = 0; r < 8; ++r) {
      float inv = 1.0f / ssum[mb + r];
      long n = n0 + mb + r;
      otile[mb + r][d0 + c]      = acc0[r] * inv + x[((long)b * NN + n) * DD + d0 + c];
      otile[mb + r][d0 + 16 + c] = acc1[r] * inv + x[((long)b * NN + n) * DD + d0 + 16 + c];
    }
  }
  __syncthreads();
  if (tid < 16) {
    float s = 0.f, s2 = 0.f;
    for (int d = 0; d < DD; ++d) { float v = otile[tid][d]; s += v; s2 += v * v; }
    float mu = s / DD, var = s2 / DD - mu * mu;
    sstat[tid] = mu; ssum[tid] = rsqrtf(var + LN_EPS);
  }
  __syncthreads();
  for (int idx = tid; idx < 16 * DD; idx += 128) {
    int r = idx >> 7, d = idx & (DD - 1);
    float v = (otile[r][d] - sstat[r]) * ssum[r] * ln1g[d] + ln1b[d];
    v *= (float)Gb[(hb * NN + n0 + r) * DD + d];
    attn_out[((long)b * NN + n0 + r) * (DD * NH) + (long)d * NH + h] = (__bf16)v;
  }
}

// ---------------- node out: relu(attn_out @ Wout) + x -> LN2 ----------------
__global__ __launch_bounds__(128) void k_nodeout(const __bf16* attn_out, const __bf16* nowT, const float* x,
                          const float* g2, const float* b2,
                          float* h_node, __bf16* h_nodeb, __bf16* h_nodeTb) {
  long bn0 = (long)blockIdx.x * 16;
  int tid = threadIdx.x, lane = tid & 31, wv = tid >> 5;
  int c = lane & 15, mb = 8 * (lane >> 4);
  __shared__ float otile[16][DD];
  __shared__ float smu[16], srs[16];
  const __bf16* A = attn_out + (bn0 + c) * (DD * NH);
  f32x8 acc0{}, acc1{};
  int d0 = wv * 32;
  for (int kt = 0; kt < (DD * NH) / 32; ++kt) {
    int k0 = kt * 32;
    bf16x16 a = ldfrag_bf16(A, k0, lane);
    acc0 = wmma_bf16(a, ldfragB_bf16(nowT + (long)(d0 + c) * (DD * NH), k0, lane), acc0);
    acc1 = wmma_bf16(a, ldfragB_bf16(nowT + (long)(d0 + 16 + c) * (DD * NH), k0, lane), acc1);
  }
#pragma unroll
  for (int r = 0; r < 8; ++r) {
    float v0 = acc0[r] > 0.f ? acc0[r] : 0.f;
    float v1 = acc1[r] > 0.f ? acc1[r] : 0.f;
    otile[mb + r][d0 + c]      = v0 + x[(bn0 + mb + r) * DD + d0 + c];
    otile[mb + r][d0 + 16 + c] = v1 + x[(bn0 + mb + r) * DD + d0 + 16 + c];
  }
  __syncthreads();
  if (tid < 16) {
    float s = 0.f, s2 = 0.f;
    for (int d = 0; d < DD; ++d) { float v = otile[tid][d]; s += v; s2 += v * v; }
    float mu = s / DD, var = s2 / DD - mu * mu;
    smu[tid] = mu; srs[tid] = rsqrtf(var + LN_EPS);
  }
  __syncthreads();
  for (int idx = tid; idx < 16 * DD; idx += 128) {
    int r = idx >> 7, d = idx & (DD - 1);
    float v = (otile[r][d] - smu[r]) * srs[r] * g2[d] + b2[d];
    long bn = bn0 + r;
    h_node[bn * DD + d] = v;
    h_nodeb[bn * DD + d] = (__bf16)v;
    long bch = bn / NN, n = bn % NN;
    h_nodeTb[(bch * DD + d) * NN + n] = (__bf16)v;
  }
}

// ---------------- edge mixing: relu((coadj@ef @ eo1) @ eo2) + ef -> LN ----------------
__global__ __launch_bounds__(128) void k_edgemix(const float* edge_coadj, const __bf16* efT, const float* ef,
                          const __bf16* eo1T, const __bf16* eo2T,
                          const float* g, const float* bt,
                          float* edge_h, __bf16* edge_hb) {
  int e0 = blockIdx.x * 16, b = blockIdx.y;
  int tid = threadIdx.x, lane = tid & 31, wv = tid >> 5;
  int c = lane & 15, mb = 8 * (lane >> 4);
  __shared__ __bf16 s_ae[16][ED];
  __shared__ __bf16 s_t1[16][HD];
  __shared__ float otile[16][ED];
  __shared__ float smu[16], srs[16];
  {  // AE = coadj @ ef
    const float* A = edge_coadj + ((long)b * NE + e0 + c) * NE;
    f32x8 acc{};
    for (int kt = 0; kt < NE / 32; ++kt) {
      int k0 = kt * 32;
      bf16x16 a = ldfrag_f32(A, k0, lane);
      acc = wmma_bf16(a, ldfragB_bf16(efT + ((long)b * ED + wv * 16 + c) * NE, k0, lane), acc);
    }
#pragma unroll
    for (int r = 0; r < 8; ++r) s_ae[mb + r][wv * 16 + c] = (__bf16)acc[r];
  }
  __syncthreads();
  {  // t1 = AE @ eo1   (no relu per reference)
    f32x8 acc{};
    for (int kt = 0; kt < ED / 32; ++kt) {
      int k0 = kt * 32;
      bf16x16 a = ldfrag_bf16(&s_ae[c][0], k0, lane);
      acc = wmma_bf16(a, ldfragB_bf16(eo1T + (long)(wv * 16 + c) * ED, k0, lane), acc);
    }
#pragma unroll
    for (int r = 0; r < 8; ++r) s_t1[mb + r][wv * 16 + c] = (__bf16)acc[r];
  }
  __syncthreads();
  {  // t2 = t1 @ eo2 ; relu ; + ef
    f32x8 acc{};
    for (int kt = 0; kt < HD / 32; ++kt) {
      int k0 = kt * 32;
      bf16x16 a = ldfrag_bf16(&s_t1[c][0], k0, lane);
      acc = wmma_bf16(a, ldfragB_bf16(eo2T + (long)(wv * 16 + c) * HD, k0, lane), acc);
    }
#pragma unroll
    for (int r = 0; r < 8; ++r) {
      float v = acc[r] > 0.f ? acc[r] : 0.f;
      otile[mb + r][wv * 16 + c] = v + ef[((long)b * NE + e0 + mb + r) * ED + wv * 16 + c];
    }
  }
  __syncthreads();
  if (tid < 16) {
    float s = 0.f, s2 = 0.f;
    for (int d = 0; d < ED; ++d) { float v = otile[tid][d]; s += v; s2 += v * v; }
    float mu = s / ED, var = s2 / ED - mu * mu;
    smu[tid] = mu; srs[tid] = rsqrtf(var + LN_EPS);
  }
  __syncthreads();
  for (int idx = tid; idx < 16 * ED; idx += 128) {
    int r = idx >> 6, d = idx & (ED - 1);
    float v = (otile[r][d] - smu[r]) * srs[r] * g[d] + bt[d];
    long e = (long)b * NE + e0 + r;
    edge_h[e * ED + d] = v;
    edge_hb[e * ED + d] = (__bf16)v;
  }
}

// ---------------- edge update: [edge_h | inc^T @ h_node] MLP -> LN3 ----------------
__global__ __launch_bounds__(128) void k_edgeupd(const float* incidence, const __bf16* h_nodeTb,
                          const float* edge_h, const __bf16* edge_hb,
                          const __bf16* ue1T, const __bf16* ue2T,
                          const float* g, const float* bt,
                          float* out_edge, __bf16* edge_newTb) {
  int e0 = blockIdx.x * 16, b = blockIdx.y;
  int tid = threadIdx.x, lane = tid & 31, wv = tid >> 5;
  int c = lane & 15, mb = 8 * (lane >> 4);
  __shared__ __bf16 s_cat[16][ED + DD];
  __shared__ __bf16 s_m1[16][HD];
  __shared__ float otile[16][ED];
  __shared__ float smu[16], srs[16];
  for (int idx = tid; idx < 16 * ED; idx += 128) {
    int r = idx >> 6, d = idx & (ED - 1);
    s_cat[r][d] = edge_hb[((long)b * NE + e0 + r) * ED + d];
  }
  {  // agg_node tile (16 x 128)
    const float* Abase = incidence + (long)b * NN * NE + (e0 + c);
    int d0 = wv * 32;
    f32x8 acc0{}, acc1{};
    for (int kt = 0; kt < NN / 32; ++kt) {
      int k0 = kt * 32;
      bf16x16 a = ldfrag_f32_strided(Abase, NE, k0, lane);
      acc0 = wmma_bf16(a, ldfragB_bf16(h_nodeTb + ((long)b * DD + d0 + c) * NN, k0, lane), acc0);
      acc1 = wmma_bf16(a, ldfragB_bf16(h_nodeTb + ((long)b * DD + d0 + 16 + c) * NN, k0, lane), acc1);
    }
#pragma unroll
    for (int r = 0; r < 8; ++r) {
      s_cat[mb + r][ED + d0 + c] = (__bf16)acc0[r];
      s_cat[mb + r][ED + d0 + 16 + c] = (__bf16)acc1[r];
    }
  }
  __syncthreads();
  {  // m1 = relu(cat @ ue1)
    f32x8 acc{};
    for (int kt = 0; kt < (ED + DD) / 32; ++kt) {
      int k0 = kt * 32;
      bf16x16 a = ldfrag_bf16(&s_cat[c][0], k0, lane);
      acc = wmma_bf16(a, ldfragB_bf16(ue1T + (long)(wv * 16 + c) * (ED + DD), k0, lane), acc);
    }
#pragma unroll
    for (int r = 0; r < 8; ++r) { float v = acc[r]; s_m1[mb + r][wv * 16 + c] = (__bf16)(v > 0.f ? v : 0.f); }
  }
  __syncthreads();
  {  // m2 = m1 @ ue2 ; + edge_h
    f32x8 acc{};
    for (int kt = 0; kt < HD / 32; ++kt) {
      int k0 = kt * 32;
      bf16x16 a = ldfrag_bf16(&s_m1[c][0], k0, lane);
      acc = wmma_bf16(a, ldfragB_bf16(ue2T + (long)(wv * 16 + c) * HD, k0, lane), acc);
    }
#pragma unroll
    for (int r = 0; r < 8; ++r)
      otile[mb + r][wv * 16 + c] = acc[r] + edge_h[((long)b * NE + e0 + mb + r) * ED + wv * 16 + c];
  }
  __syncthreads();
  if (tid < 16) {
    float s = 0.f, s2 = 0.f;
    for (int d = 0; d < ED; ++d) { float v = otile[tid][d]; s += v; s2 += v * v; }
    float mu = s / ED, var = s2 / ED - mu * mu;
    smu[tid] = mu; srs[tid] = rsqrtf(var + LN_EPS);
  }
  __syncthreads();
  for (int idx = tid; idx < 16 * ED; idx += 128) {
    int r = idx >> 6, d = idx & (ED - 1);
    float v = (otile[r][d] - smu[r]) * srs[r] * g[d] + bt[d];
    out_edge[((long)b * NE + e0 + r) * ED + d] = v;
    edge_newTb[((long)b * ED + d) * NE + e0 + r] = (__bf16)v;
  }
}

// ---------------- node update: [h_node | inc @ edge_new] MLP -> LN3 ----------------
__global__ __launch_bounds__(128) void k_nodeupd(const float* incidence, const __bf16* edge_newTb,
                          const float* h_node, const __bf16* h_nodeb,
                          const __bf16* un1T, const __bf16* un2T,
                          const float* g, const float* bt,
                          float* out_node, __bf16* node_newb) {
  int n0 = blockIdx.x * 16, b = blockIdx.y;
  int tid = threadIdx.x, lane = tid & 31, wv = tid >> 5;
  int c = lane & 15, mb = 8 * (lane >> 4);
  __shared__ __bf16 s_cat[16][DD + ED];
  __shared__ __bf16 s_m1[16][HD];
  __shared__ float otile[16][DD];
  __shared__ float smu[16], srs[16];
  for (int idx = tid; idx < 16 * DD; idx += 128) {
    int r = idx >> 7, d = idx & (DD - 1);
    s_cat[r][d] = h_nodeb[((long)b * NN + n0 + r) * DD + d];
  }
  {  // agg_edge (16 x 64)
    const float* A = incidence + ((long)b * NN + n0 + c) * NE;
    f32x8 acc{};
    for (int kt = 0; kt < NE / 32; ++kt) {
      int k0 = kt * 32;
      bf16x16 a = ldfrag_f32(A, k0, lane);
      acc = wmma_bf16(a, ldfragB_bf16(edge_newTb + ((long)b * ED + wv * 16 + c) * NE, k0, lane), acc);
    }
#pragma unroll
    for (int r = 0; r < 8; ++r) s_cat[mb + r][DD + wv * 16 + c] = (__bf16)acc[r];
  }
  __syncthreads();
  {  // m1 = relu(cat @ un1)
    f32x8 acc{};
    for (int kt = 0; kt < (DD + ED) / 32; ++kt) {
      int k0 = kt * 32;
      bf16x16 a = ldfrag_bf16(&s_cat[c][0], k0, lane);
      acc = wmma_bf16(a, ldfragB_bf16(un1T + (long)(wv * 16 + c) * (DD + ED), k0, lane), acc);
    }
#pragma unroll
    for (int r = 0; r < 8; ++r) { float v = acc[r]; s_m1[mb + r][wv * 16 + c] = (__bf16)(v > 0.f ? v : 0.f); }
  }
  __syncthreads();
  {  // m2 = m1 @ un2 (16x128) ; + h_node
    int d0 = wv * 32;
    f32x8 acc0{}, acc1{};
    for (int kt = 0; kt < HD / 32; ++kt) {
      int k0 = kt * 32;
      bf16x16 a = ldfrag_bf16(&s_m1[c][0], k0, lane);
      acc0 = wmma_bf16(a, ldfragB_bf16(un2T + (long)(d0 + c) * HD, k0, lane), acc0);
      acc1 = wmma_bf16(a, ldfragB_bf16(un2T + (long)(d0 + 16 + c) * HD, k0, lane), acc1);
    }
#pragma unroll
    for (int r = 0; r < 8; ++r) {
      long n = n0 + mb + r;
      otile[mb + r][d0 + c]      = acc0[r] + h_node[((long)b * NN + n) * DD + d0 + c];
      otile[mb + r][d0 + 16 + c] = acc1[r] + h_node[((long)b * NN + n) * DD + d0 + 16 + c];
    }
  }
  __syncthreads();
  if (tid < 16) {
    float s = 0.f, s2 = 0.f;
    for (int d = 0; d < DD; ++d) { float v = otile[tid][d]; s += v; s2 += v * v; }
    float mu = s / DD, var = s2 / DD - mu * mu;
    smu[tid] = mu; srs[tid] = rsqrtf(var + LN_EPS);
  }
  __syncthreads();
  for (int idx = tid; idx < 16 * DD; idx += 128) {
    int r = idx >> 7, d = idx & (DD - 1);
    float v = (otile[r][d] - smu[r]) * srs[r] * g[d] + bt[d];
    long bn = (long)b * NN + n0 + r;
    out_node[bn * DD + d] = v;
    node_newb[bn * DD + d] = (__bf16)v;
  }
}

// ---------------- coordinate MLP per edge (WMMA over 16-edge tiles) ----------------
__global__ __launch_bounds__(256) void k_coord(const __bf16* node_newb, const float* cf,
                       const int* fsrc, const int* fdst,
                       const __bf16* cm1Tp, const float* cm2, float* wbuf) {
  int lane = threadIdx.x & 31, wv = threadIdx.x >> 5;
  int c = lane & 15, mb = 8 * (lane >> 4);
  int e0 = (blockIdx.x * 8 + wv) * 16;
  __shared__ float s_h[8][16][HD];   // 32 KB
  __shared__ float s_dist[8][16];
  if (lane < 16) {
    int e = e0 + lane;
    int s = fsrc[e], t = fdst[e];
    float dx = cf[s * 3 + 0] - cf[t * 3 + 0];
    float dy = cf[s * 3 + 1] - cf[t * 3 + 1];
    float dz = cf[s * 3 + 2] - cf[t * 3 + 2];
    s_dist[wv][lane] = sqrtf(dx * dx + dy * dy + dz * dz + 1e-8f);
  }
  __syncthreads();
  int erow = e0 + c;
  const __bf16* ni = node_newb + (long)fsrc[erow] * DD;
  const __bf16* nj = node_newb + (long)fdst[erow] * DD;
  f32x8 acc[4] = {};
  for (int kt = 0; kt < 9; ++kt) {
    int k0 = kt * 32;
    bf16x16 a;
    if (kt < 4) a = ldfrag_bf16(ni, k0, lane);
    else if (kt < 8) a = ldfrag_bf16(nj, k0 - DD, lane);
    else {
#pragma unroll
      for (int e2 = 0; e2 < 16; ++e2) a[e2] = (__bf16)0.0f;
      if (frag_klo(lane) == 0) a[0] = (__bf16)s_dist[wv][c];  // K==256 -> dist
    }
#pragma unroll
    for (int ct = 0; ct < 4; ++ct)
      acc[ct] = wmma_bf16(a, ldfragB_bf16(cm1Tp + (long)(ct * 16 + c) * 288, k0, lane), acc[ct]);
  }
#pragma unroll
  for (int ct = 0; ct < 4; ++ct)
#pragma unroll
    for (int r = 0; r < 8; ++r) {
      float v = acc[ct][r];
      s_h[wv][mb + r][ct * 16 + c] = v / (1.f + __expf(-v));  // silu
    }
  __syncthreads();
  if (lane < 16) {
    float w = 0.f;
    for (int j = 0; j < HD; ++j) w += s_h[wv][lane][j] * cm2[j];
    wbuf[e0 + lane] = w;
  }
}

// deterministic per-node scatter: node bn owns edges [bn*K, bn*K+K)
__global__ void k_coords_out(const float* cf, const float* wbuf, const int* fsrc, const int* fdst,
                             const float* mask, float* out3, int Kdeg) {
  int bn = blockIdx.x * 256 + threadIdx.x;
  if (bn >= BB * NN) return;
  float a0 = 0.f, a1 = 0.f, a2 = 0.f;
  for (int kk = 0; kk < Kdeg; ++kk) {
    int e = bn * Kdeg + kk;
    int s = fsrc[e], t = fdst[e];
    float w = wbuf[e];
    a0 += (cf[s * 3 + 0] - cf[t * 3 + 0]) * w;
    a1 += (cf[s * 3 + 1] - cf[t * 3 + 1]) * w;
    a2 += (cf[s * 3 + 2] - cf[t * 3 + 2]) * w;
  }
  float inv = 1.0f / (Kdeg + 1.0f);
  float mk = mask[bn];
  out3[bn * 3 + 0] = (cf[bn * 3 + 0] + a0 * inv) * mk;
  out3[bn * 3 + 1] = (cf[bn * 3 + 1] + a1 * inv) * mk;
  out3[bn * 3 + 2] = (cf[bn * 3 + 2] + a2 * inv) * mk;
}

// ---------------- host launcher ----------------
extern "C" void kernel_launch(void* const* d_in, const int* in_sizes, int n_in,
                              void* d_out, int out_size, void* d_ws, size_t ws_size,
                              hipStream_t stream) {
  const float* x          = (const float*)d_in[0];
  const float* ef         = (const float*)d_in[1];
  const float* coords     = (const float*)d_in[2];
  const float* node_adj   = (const float*)d_in[3];
  const float* edge_coadj = (const float*)d_in[4];
  const float* incidence  = (const float*)d_in[5];
  const float* atom_mask  = (const float*)d_in[6];
  const int*   fsrc       = (const int*)d_in[7];
  const int*   fdst       = (const int*)d_in[8];
  const float* W_key      = (const float*)d_in[9];
  const float* W_query    = (const float*)d_in[10];
  const float* W_value    = (const float*)d_in[11];
  const float* node_bias  = (const float*)d_in[12];
  const float* node_gate  = (const float*)d_in[13];
  const float* node_out_w = (const float*)d_in[14];
  const float* ln1g = (const float*)d_in[15];
  const float* ln1b = (const float*)d_in[16];
  const float* ln2g = (const float*)d_in[17];
  const float* ln2b = (const float*)d_in[18];
  const float* ln3g = (const float*)d_in[19];
  const float* ln3b = (const float*)d_in[20];
  const float* eo1  = (const float*)d_in[21];
  const float* eo2  = (const float*)d_in[22];
  const float* eln2g = (const float*)d_in[23];
  const float* eln2b = (const float*)d_in[24];
  const float* eln3g = (const float*)d_in[25];
  const float* eln3b = (const float*)d_in[26];
  const float* ue1  = (const float*)d_in[27];
  const float* ue2  = (const float*)d_in[28];
  const float* un1  = (const float*)d_in[29];
  const float* un2  = (const float*)d_in[30];
  const float* cm1  = (const float*)d_in[31];
  const float* cm2  = (const float*)d_in[32];
  const int Kdeg = in_sizes[7] / (BB * NN);

  char* wp = (char*)d_ws;
  auto alloc = [&](size_t bytes) -> void* {
    void* p = (void*)wp;
    wp += (bytes + 255) & ~(size_t)255;
    return p;
  };
  __bf16* xb   = (__bf16*)alloc(sizeof(__bf16) * (size_t)BB * NN * DD);
  __bf16* WqT  = (__bf16*)alloc(sizeof(__bf16) * (size_t)NH * HD * DD);
  __bf16* WkT  = (__bf16*)alloc(sizeof(__bf16) * (size_t)NH * HD * DD);
  __bf16* WvT  = (__bf16*)alloc(sizeof(__bf16) * (size_t)NH * DD * DD);
  __bf16* WgT  = (__bf16*)alloc(sizeof(__bf16) * (size_t)NH * DD * DD);
  __bf16* nowT = (__bf16*)alloc(sizeof(__bf16) * (size_t)DD * NH * DD);
  __bf16* eo1T = (__bf16*)alloc(sizeof(__bf16) * HD * ED);
  __bf16* eo2T = (__bf16*)alloc(sizeof(__bf16) * ED * HD);
  __bf16* ue1T = (__bf16*)alloc(sizeof(__bf16) * HD * (ED + DD));
  __bf16* ue2T = (__bf16*)alloc(sizeof(__bf16) * ED * HD);
  __bf16* un1T = (__bf16*)alloc(sizeof(__bf16) * HD * (ED + DD));
  __bf16* un2T = (__bf16*)alloc(sizeof(__bf16) * DD * HD);
  __bf16* cm1Tp = (__bf16*)alloc(sizeof(__bf16) * HD * 288);
  __bf16* efT  = (__bf16*)alloc(sizeof(__bf16) * (size_t)BB * ED * NE);
  __bf16* Qb   = (__bf16*)alloc(sizeof(__bf16) * (size_t)NH * BB * NN * HD);
  __bf16* Kb   = (__bf16*)alloc(sizeof(__bf16) * (size_t)NH * BB * NN * HD);
  __bf16* Vt   = (__bf16*)alloc(sizeof(__bf16) * (size_t)NH * BB * DD * NN);
  __bf16* Gb   = (__bf16*)alloc(sizeof(__bf16) * (size_t)NH * BB * NN * DD);
  float*  Bbuf = (float*)alloc(sizeof(float) * (size_t)NH * BB * NN);
  __bf16* attn_out = (__bf16*)alloc(sizeof(__bf16) * (size_t)BB * NN * DD * NH);
  float*  h_node   = (float*)alloc(sizeof(float) * (size_t)BB * NN * DD);
  __bf16* h_nodeb  = (__bf16*)alloc(sizeof(__bf16) * (size_t)BB * NN * DD);
  __bf16* h_nodeTb = (__bf16*)alloc(sizeof(__bf16) * (size_t)BB * DD * NN);
  float*  edge_h   = (float*)alloc(sizeof(float) * (size_t)BB * NE * ED);
  __bf16* edge_hb  = (__bf16*)alloc(sizeof(__bf16) * (size_t)BB * NE * ED);
  __bf16* edge_newTb = (__bf16*)alloc(sizeof(__bf16) * (size_t)BB * ED * NE);
  __bf16* node_newb  = (__bf16*)alloc(sizeof(__bf16) * (size_t)BB * NN * DD);
  float*  wbuf = (float*)alloc(sizeof(float) * (size_t)BB * NN * Kdeg);

  float* out_node  = (float*)d_out;
  float* out_edge  = out_node + (size_t)BB * NN * DD;
  float* out_coord = out_edge + (size_t)BB * NE * ED;

  auto T = [&](__bf16* dst, const float* src, int R, int C, int Rpad) {
    int total = C * Rpad;
    k_transpose<<<(total + 255) / 256, 256, 0, stream>>>(dst, src, R, C, Rpad);
  };

  // stage 0: conversions / transposes
  k_cvt<<<((BB * NN * DD) + 255) / 256, 256, 0, stream>>>(xb, x, BB * NN * DD);
  for (int h = 0; h < NH; ++h) {
    T(WqT + (size_t)h * HD * DD, W_query + (size_t)h * DD * HD, DD, HD, DD);
    T(WkT + (size_t)h * HD * DD, W_key   + (size_t)h * DD * HD, DD, HD, DD);
    T(WvT + (size_t)h * DD * DD, W_value + (size_t)h * DD * DD, DD, DD, DD);
    T(WgT + (size_t)h * DD * DD, node_gate + (size_t)h * DD * DD, DD, DD, DD);
  }
  T(nowT, node_out_w, NH * DD, DD, NH * DD);
  T(eo1T, eo1, ED, HD, ED);
  T(eo2T, eo2, HD, ED, HD);
  T(ue1T, ue1, ED + DD, HD, ED + DD);
  T(ue2T, ue2, HD, ED, HD);
  T(un1T, un1, ED + DD, HD, ED + DD);
  T(un2T, un2, HD, DD, HD);
  T(cm1Tp, cm1, 2 * DD + 1, HD, 288);
  for (int b = 0; b < BB; ++b)
    T(efT + (size_t)b * ED * NE, ef + (size_t)b * NE * ED, NE, ED, NE);
  k_bias<<<(NH * BB * NN + 255) / 256, 256, 0, stream>>>(Bbuf, x, node_bias);

  // stage 1: projections
  k_proj<<<dim3(NN / 16, BB, NH), 128, 0, stream>>>(xb, WqT, WkT, WvT, WgT, Qb, Kb, Vt, Gb);
  // stage 2: fused attention
  k_attn<<<dim3(NN / 16, BB, NH), 128, 0, stream>>>(Qb, Kb, Vt, Gb, Bbuf, node_adj, x, ln1g, ln1b, attn_out);
  // stage 3: node out + LN2
  k_nodeout<<<dim3((BB * NN) / 16), 128, 0, stream>>>(attn_out, nowT, x, ln2g, ln2b, h_node, h_nodeb, h_nodeTb);
  // stage 4: edge mixing
  k_edgemix<<<dim3(NE / 16, BB), 128, 0, stream>>>(edge_coadj, efT, ef, eo1T, eo2T, eln2g, eln2b, edge_h, edge_hb);
  // stage 5: edge update -> out_edge
  k_edgeupd<<<dim3(NE / 16, BB), 128, 0, stream>>>(incidence, h_nodeTb, edge_h, edge_hb, ue1T, ue2T,
                                                   eln3g, eln3b, out_edge, edge_newTb);
  // stage 6: node update -> out_node
  k_nodeupd<<<dim3(NN / 16, BB), 128, 0, stream>>>(incidence, edge_newTb, h_node, h_nodeb, un1T, un2T,
                                                   ln3g, ln3b, out_node, node_newb);
  // stage 7: coord MLP per edge, then deterministic per-node gather
  k_coord<<<dim3((BB * NN * Kdeg) / 128), 256, 0, stream>>>(node_newb, coords, fsrc, fdst, cm1Tp, cm2, wbuf);
  k_coords_out<<<dim3((BB * NN + 255) / 256), 256, 0, stream>>>(coords, wbuf, fsrc, fdst, atom_mask,
                                                                out_coord, Kdeg);
  (void)n_in; (void)out_size; (void)ws_size; (void)in_sizes;
}
